// EIRTBlock_26628797235476
// MI455X (gfx1250) — matrix operand order repaired and checked by reading, not compiled, not run on hardware
//
#include <hip/hip_runtime.h>
#include <hip/hip_bf16.h>
#include <stdint.h>

// Problem constants (match reference)
#define B_    4
#define INDIM 64
#define C_    128
#define H_    64
#define W_    64
#define HW    (H_ * W_)
#define EPSLN 1e-5f

typedef __attribute__((ext_vector_type(16))) _Float16 v16h;
typedef __attribute__((ext_vector_type(8)))  float    v8f;

union Frag { uint4 q[2]; v16h v; };

__device__ __forceinline__ float lrelu_f(float x) { return x >= 0.f ? x : 0.01f * x; }

// CDNA5 async global->LDS DMA (ASYNCcnt), no VGPR round trip.
// Generic pointers to __shared__ carry the LDS byte offset in bits [31:0]
// (shared-aperture mapping), which the instruction takes as the LDS address.
__device__ __forceinline__ void async_ld_b128(void* lds_ptr, const void* gptr)
{
    unsigned lds_off = (unsigned)(uintptr_t)lds_ptr;
    asm volatile("global_load_async_to_lds_b128 %0, %1, off"
                 :: "v"(lds_off), "v"(gptr) : "memory");
}
__device__ __forceinline__ void wait_async0()
{
    asm volatile("s_wait_asynccnt 0x0" ::: "memory");
}

// ---------------------------------------------------------------------------
// Direct 3x3 convolution (pad=1), NCHW fp32 -> NCHW fp32.  Small FLOP share.
// ---------------------------------------------------------------------------
__global__ __launch_bounds__(256)
void conv3x3_kernel(const float* __restrict__ x, const float* __restrict__ wgt,
                    const float* __restrict__ bias, float* __restrict__ y,
                    int Cin, int Cout)
{
    int idx = blockIdx.x * 256 + threadIdx.x;
    int p  = idx % HW;
    int co = (idx / HW) % Cout;
    int b  = idx / (HW * Cout);
    int h = p / W_, w = p % W_;
    float acc = bias[co];
    const float* wrow = wgt + (size_t)co * Cin * 9;
    for (int ci = 0; ci < Cin; ++ci) {
        const float* xin = x + ((size_t)b * Cin + ci) * HW;
        const float* wk  = wrow + ci * 9;
#pragma unroll
        for (int kh = 0; kh < 3; ++kh) {
            int ih = h + kh - 1;
            if (ih < 0 || ih >= H_) continue;
#pragma unroll
            for (int kw = 0; kw < 3; ++kw) {
                int iw = w + kw - 1;
                if (iw < 0 || iw >= W_) continue;
                acc += wk[kh * 3 + kw] * xin[ih * W_ + iw];
            }
        }
    }
    y[idx] = acc;
}

// ---------------------------------------------------------------------------
// LayerNorm reduction: one block per sample (C*HW contiguous floats,
// layout-agnostic for mean/var). stats[b] = {mean, rstd}.
// ---------------------------------------------------------------------------
__global__ __launch_bounds__(512)
void ln_reduce_kernel(const float* __restrict__ x, float* __restrict__ stats)
{
    __shared__ float s1[512], s2[512];
    int b = blockIdx.x;
    const float* xb = x + (size_t)b * C_ * HW;
    float s = 0.f, ss = 0.f;
    for (int i = threadIdx.x; i < C_ * HW; i += 512) {
        float v = xb[i]; s += v; ss += v * v;
    }
    s1[threadIdx.x] = s; s2[threadIdx.x] = ss;
    __syncthreads();
    for (int off = 256; off > 0; off >>= 1) {
        if ((int)threadIdx.x < off) {
            s1[threadIdx.x] += s1[threadIdx.x + off];
            s2[threadIdx.x] += s2[threadIdx.x + off];
        }
        __syncthreads();
    }
    if (threadIdx.x == 0) {
        float inv = 1.f / (float)(C_ * HW);
        float mean = s1[0] * inv;
        float var  = s2[0] * inv - mean * mean;
        stats[b * 2 + 0] = mean;
        stats[b * 2 + 1] = rsqrtf(var + EPSLN);
    }
}

// LN apply + affine + LeakyReLU from NCHW fp32 input.
// Optional outputs: fp32 NCHW, fp16 NCHW, fp16 token-major [b][p][c].
__global__ __launch_bounds__(256)
void ln_apply_nchw_kernel(const float* __restrict__ x, const float* __restrict__ g,
                          const float* __restrict__ be, const float* __restrict__ stats,
                          float* __restrict__ outF, _Float16* __restrict__ outHn,
                          _Float16* __restrict__ outHt)
{
    int idx = blockIdx.x * 256 + threadIdx.x;
    int p = idx % HW;
    int c = (idx / HW) % C_;
    int b = idx / (HW * C_);
    float mean = stats[b * 2], rstd = stats[b * 2 + 1];
    float v = (x[idx] - mean) * rstd * g[c * HW + p] + be[c * HW + p];
    v = lrelu_f(v);
    if (outF)  outF[idx] = v;
    if (outHn) outHn[idx] = (_Float16)v;
    if (outHt) outHt[((size_t)b * HW + p) * C_ + c] = (_Float16)v;
}

// LN apply + affine + LeakyReLU from token-major fp32 input [b][p][d],
// writes NCHW fp32 (feeds the 3x3 conv).
__global__ __launch_bounds__(256)
void ln_apply_tok_kernel(const float* __restrict__ x, const float* __restrict__ g,
                         const float* __restrict__ be, const float* __restrict__ stats,
                         float* __restrict__ outFn)
{
    int idx = blockIdx.x * 256 + threadIdx.x;
    int d = idx % C_;
    int p = (idx / C_) % HW;
    int b = idx / (C_ * HW);
    float mean = stats[b * 2], rstd = stats[b * 2 + 1];
    float v = (x[idx] - mean) * rstd * g[d * HW + p] + be[d * HW + p];
    outFn[((size_t)b * C_ + d) * HW + p] = lrelu_f(v);
}

__global__ __launch_bounds__(256)
void cvt_f32_f16_kernel(const float* __restrict__ x, _Float16* __restrict__ y, int n)
{
    int i = blockIdx.x * 256 + threadIdx.x;
    if (i < n) y[i] = (_Float16)x[i];
}

__global__ __launch_bounds__(256)
void mul_h_kernel(const _Float16* __restrict__ a, const _Float16* __restrict__ b,
                  _Float16* __restrict__ o)
{
    int i = blockIdx.x * 256 + threadIdx.x;
    o[i] = (_Float16)((float)a[i] * (float)b[i]);
}

// ---------------------------------------------------------------------------
// Generic WMMA GEMM:  out[r][n] = act( sum_k A[r][k] * Wt[n][k] + bias[n] )
// A: row-major M x K fp16.  Wt: row-major N x K fp16 (k-contiguous per output
// column = exactly the B-fragment layout).  Workgroup tile 128(M) x 64(N),
// BK=32, 8 waves each owning 32x32 (4 WMMAs per chunk).
// Double-buffered LDS fed by global_load_async_to_lds_b128 (ASYNCcnt): the
// next K-tile streams from HBM/L2 while the current one feeds the WMMAs.
// LDS pitch 40 halfwords = 80B: multiple of 16B (aligned b128 ops) and
// conflict-free (20*lane mod 64 permutes over 16 lanes).
// out_mode: 0 = row-major (fp32/fp16), 1 = fp32 NCHW (final conv1x1),
//           2 = fp16 [b][m][c] transpose of the (b,c)-row x m-col result
//               (emits f_i_3 directly in the attention K-tile layout).
// ---------------------------------------------------------------------------
__global__ __launch_bounds__(256)
void gemm_wmma_kernel(const _Float16* __restrict__ A, const _Float16* __restrict__ Wt,
                      const float* __restrict__ bias, float* __restrict__ outF,
                      _Float16* __restrict__ outH, int M, int N, int K,
                      int relu, int out_mode)
{
    __shared__ _Float16 As[2][128 * 40];
    __shared__ _Float16 Bs[2][64 * 40];
    const int tid  = threadIdx.x;
    const int lane = tid & 31, wave = tid >> 5;
    const int hx   = lane >> 4, l16 = lane & 15;   // half-wave, lane-in-half
    const int m0 = blockIdx.y * 128, n0 = blockIdx.x * 64;
    const int wm = (wave >> 1) * 32;   // wave M offset within tile
    const int wn = (wave & 1) * 32;    // wave N offset within tile

    auto stage = [&](int buf, int k0) {
#pragma unroll
        for (int j = 0; j < 2; ++j) {             // A: 128x32, 2 b128/thread
            int u = tid + 256 * j;
            int row = u >> 2, kk = (u & 3) * 8;
            async_ld_b128(&As[buf][row * 40 + kk],
                          &A[(size_t)(m0 + row) * K + k0 + kk]);
        }
        int row = tid >> 2, kk = (tid & 3) * 8;   // B: 64x32, 1 b128/thread
        async_ld_b128(&Bs[buf][row * 40 + kk],
                      &Wt[(size_t)(n0 + row) * K + k0 + kk]);
    };

    v8f acc[2][2] = {};
    stage(0, 0);
    wait_async0();
    __syncthreads();
    int cur = 0;
    for (int k0 = 0; k0 < K; k0 += 32) {
        if (k0 + 32 < K) stage(cur ^ 1, k0 + 32);      // prefetch next tile
        if (k0 + 64 < K)                               // warm L2 two ahead
            __builtin_prefetch(&A[(size_t)(m0 + (tid >> 2)) * K + k0 + 64], 0, 1);

        Frag aF[2], bF[2];
#pragma unroll
        for (int i = 0; i < 2; ++i) {
            int r = wm + i * 16 + l16;   // A frag: runs at k = hx*8 and 16+hx*8
            aF[i].q[0] = *reinterpret_cast<const uint4*>(&As[cur][r * 40 + hx * 8]);
            aF[i].q[1] = *reinterpret_cast<const uint4*>(&As[cur][r * 40 + 16 + hx * 8]);
            int n = wn + i * 16 + l16;   // B frag: runs at k = hx*8 and 16+hx*8
            bF[i].q[0] = *reinterpret_cast<const uint4*>(&Bs[cur][n * 40 + hx * 8]);
            bF[i].q[1] = *reinterpret_cast<const uint4*>(&Bs[cur][n * 40 + 16 + hx * 8]);
        }
#pragma unroll
        for (int i = 0; i < 2; ++i)
#pragma unroll
            for (int jn = 0; jn < 2; ++jn)
                acc[i][jn] = __builtin_amdgcn_wmma_f32_16x16x32_f16(
                    false, aF[i].v, false, bF[jn].v, (short)0, acc[i][jn], false, false);

        wait_async0();      // next tile landed in LDS
        __syncthreads();    // all waves done with cur, next buffer valid
        cur ^= 1;
    }

    // Epilogue: D layout is row = j + 8*hx, col = l16 within each 16x16 tile.
#pragma unroll
    for (int i = 0; i < 2; ++i)
#pragma unroll
        for (int jn = 0; jn < 2; ++jn) {
            int cg = n0 + wn + jn * 16 + l16;
            float bv = bias[cg];
#pragma unroll
            for (int j = 0; j < 8; ++j) {
                int rg = m0 + wm + i * 16 + j + 8 * hx;
                float v = acc[i][jn][j] + bv;
                if (relu) v = lrelu_f(v);
                if (out_mode == 0) {
                    if (outF) outF[(size_t)rg * N + cg] = v;
                    if (outH) outH[(size_t)rg * N + cg] = (_Float16)v;
                } else if (out_mode == 1) {
                    int b = rg / HW, p = rg % HW;
                    outF[((size_t)b * N + cg) * HW + p] = v;
                } else {
                    // rows are (b,c) with C_=128; emit [b][m][c]
                    outH[(((size_t)(rg >> 7)) * HW + cg) * C_ + (rg & 127)] =
                        (_Float16)v;
                }
            }
        }
}

// ---------------------------------------------------------------------------
// Flash attention (unscaled, softmax over keys), d = C_ = 128.
//   Q   = f_e_4 token-major [b][n][c]  (A fragments straight from global)
//   KtT = f_i_3 token-major [b][m][c]  (async b128 copy -> LDS [m][c])
//   V   = f_e_1 NCHW       [b*C+c][m]  (async b128 copy -> LDS [c][m])
//   O  -> token-major [b][n][c]
// Streams keys in tiles of 64: the 256MB score matrix never touches HBM
// (would cost ~23us at 23.3 TB/s, dominating the whole block).
// K-tiles are double-buffered via global_load_async_to_lds_b128, prefetched
// one iteration ahead; the V-tile load overlaps the QK^T + softmax phase.
// Online softmax reduces with shfl_xor masks 1..8: lanes 0-15/16-31 reduce
// independently, matching the D layout (VGPR j <-> row j + 8*half).
// Static LDS = 2*17408 + 18432 + 9216 = 62.5 KB.
// ---------------------------------------------------------------------------
__global__ __launch_bounds__(128)
void flash_attn_kernel(const _Float16* __restrict__ Q, const _Float16* __restrict__ KtT,
                       const _Float16* __restrict__ V, _Float16* __restrict__ O)
{
    __shared__ _Float16 Ks[2][64 * 136];  // [m_local][c], pitch 272B (17*16B)
    __shared__ _Float16 Vs[128 * 72];     // [c][m_local], pitch 144B (9*16B)
    __shared__ _Float16 Ps[4 * 16 * 72];  // per-wave P scratch, D->A relayout
    const int tid  = threadIdx.x;
    const int lane = tid & 31, wave = tid >> 5;
    const int hx   = lane >> 4, l16 = lane & 15;
    const int b  = blockIdx.y;
    const int q0 = blockIdx.x * 64 + wave * 16;

    auto stageK = [&](int buf, int m0) {    // 64 x 128 h = 1024 b128 chunks
        for (int i4 = tid; i4 < 1024; i4 += 128) {
            int mL = i4 >> 4, cc = (i4 & 15) * 8;
            async_ld_b128(&Ks[buf][mL * 136 + cc],
                          &KtT[((size_t)b * HW + m0 + mL) * C_ + cc]);
        }
    };
    auto stageV = [&](int m0) {             // 128 x 64 h = 1024 b128 chunks
        for (int i4 = tid; i4 < 1024; i4 += 128) {
            int c = i4 >> 3, mm = (i4 & 7) * 8;
            async_ld_b128(&Vs[c * 72 + mm],
                          &V[((size_t)b * C_ + c) * HW + m0 + mm]);
        }
    };

    // Q fragments (16 rows x K=128 in 4 chunks of 32), loaded once.
    Frag qF[4];
    {
        const _Float16* qrow = Q + ((size_t)b * HW + q0 + l16) * C_;
#pragma unroll
        for (int kc = 0; kc < 4; ++kc) {
            qF[kc].q[0] = *reinterpret_cast<const uint4*>(qrow + kc * 32 + hx * 8);
            qF[kc].q[1] = *reinterpret_cast<const uint4*>(qrow + kc * 32 + 16 + hx * 8);
        }
    }

    v8f oacc[8] = {};
    float mrow[8], lrow[8];
#pragma unroll
    for (int j = 0; j < 8; ++j) { mrow[j] = -1e30f; lrow[j] = 0.f; }

    stageK(0, 0);
    wait_async0();
    __syncthreads();
    int cur = 0;
    for (int m0 = 0; m0 < HW; m0 += 64) {
        stageV(m0);                               // overlaps with QK^T phase
        if (m0 + 64 < HW) stageK(cur ^ 1, m0 + 64);

        // S = Q x K^T : 16 x 64 scores, 4 n-tiles x 4 k-chunks = 16 WMMAs.
        v8f s[4] = {};
#pragma unroll
        for (int t = 0; t < 4; ++t) {
#pragma unroll
            for (int kc = 0; kc < 4; ++kc) {
                Frag bf;
                int base = (t * 16 + l16) * 136 + kc * 32 + hx * 8;
                bf.q[0] = *reinterpret_cast<const uint4*>(&Ks[cur][base]);
                bf.q[1] = *reinterpret_cast<const uint4*>(&Ks[cur][base + 16]);
                s[t] = __builtin_amdgcn_wmma_f32_16x16x32_f16(
                    false, qF[kc].v, false, bf.v, (short)0, s[t], false, false);
            }
        }

        // Online softmax, per row j (global row = q0 + j + 8*hx).
#pragma unroll
        for (int j = 0; j < 8; ++j) {
            float mx = fmaxf(fmaxf(s[0][j], s[1][j]), fmaxf(s[2][j], s[3][j]));
#pragma unroll
            for (int msk = 1; msk < 16; msk <<= 1)
                mx = fmaxf(mx, __shfl_xor(mx, msk, 32));
            float mnew  = fmaxf(mrow[j], mx);
            float scale = __expf(mrow[j] - mnew);
            float psum = 0.f;
#pragma unroll
            for (int t = 0; t < 4; ++t) {
                float pv = __expf(s[t][j] - mnew);
                s[t][j] = pv; psum += pv;
            }
#pragma unroll
            for (int msk = 1; msk < 16; msk <<= 1)
                psum += __shfl_xor(psum, msk, 32);
            lrow[j] = lrow[j] * scale + psum;
            mrow[j] = mnew;
#pragma unroll
            for (int ot = 0; ot < 8; ++ot) oacc[ot][j] = oacc[ot][j] * scale;
        }

        // P: D layout -> per-wave LDS scratch in A layout ([row][m], 16x64).
#pragma unroll
        for (int t = 0; t < 4; ++t)
#pragma unroll
            for (int j = 0; j < 8; ++j)
                Ps[wave * (16 * 72) + (j + 8 * hx) * 72 + t * 16 + l16] = (_Float16)s[t][j];

        wait_async0();      // V tile (and next K tile) landed
        __syncthreads();    // Ps visible; Vs ready

        // O += P x V : 8 col tiles x 2 k-chunks = 16 WMMAs.
#pragma unroll
        for (int kc = 0; kc < 2; ++kc) {
            Frag aP;
            int pb = wave * (16 * 72) + l16 * 72 + kc * 32;
            aP.q[0] = *reinterpret_cast<const uint4*>(&Ps[pb + hx * 8]);
            aP.q[1] = *reinterpret_cast<const uint4*>(&Ps[pb + 16 + hx * 8]);
#pragma unroll
            for (int ot = 0; ot < 8; ++ot) {
                Frag bV;
                int vb = (ot * 16 + l16) * 72 + kc * 32 + hx * 8;
                bV.q[0] = *reinterpret_cast<const uint4*>(&Vs[vb]);
                bV.q[1] = *reinterpret_cast<const uint4*>(&Vs[vb + 16]);
                oacc[ot] = __builtin_amdgcn_wmma_f32_16x16x32_f16(
                    false, aP.v, false, bV.v, (short)0, oacc[ot], false, false);
            }
        }
        __syncthreads();    // all waves done with Vs/Ps before next overwrite
        cur ^= 1;
    }

    // Normalize and write token-major output.
#pragma unroll
    for (int ot = 0; ot < 8; ++ot)
#pragma unroll
        for (int j = 0; j < 8; ++j) {
            int r = q0 + j + 8 * hx;
            O[((size_t)b * HW + r) * C_ + ot * 16 + l16] =
                (_Float16)(oacc[ot][j] / lrow[j]);
        }
}

// ---------------------------------------------------------------------------
extern "C" void kernel_launch(void* const* d_in, const int* in_sizes, int n_in,
                              void* d_out, int out_size, void* d_ws, size_t ws_size,
                              hipStream_t stream)
{
    (void)in_sizes; (void)n_in; (void)out_size; (void)ws_size;
    const float* f_e = (const float*)d_in[0];
    const float* f_i = (const float*)d_in[1];
    const float* w1  = (const float*)d_in[2];  const float* b1  = (const float*)d_in[3];
    const float* g1  = (const float*)d_in[4];  const float* be1 = (const float*)d_in[5];
    const float* w2  = (const float*)d_in[6];  const float* b2  = (const float*)d_in[7];
    const float* g2  = (const float*)d_in[8];  const float* be2 = (const float*)d_in[9];
    const float* W3  = (const float*)d_in[10]; const float* b3  = (const float*)d_in[11];
    const float* W4  = (const float*)d_in[12]; const float* b4  = (const float*)d_in[13];
    const float* W5  = (const float*)d_in[14]; const float* b5  = (const float*)d_in[15];
    const float* w6a = (const float*)d_in[16]; const float* b6a = (const float*)d_in[17];
    const float* g6a = (const float*)d_in[18]; const float* be6a= (const float*)d_in[19];
    const float* w6b = (const float*)d_in[20]; const float* b6b = (const float*)d_in[21];
    const float* g6b = (const float*)d_in[22]; const float* be6b= (const float*)d_in[23];
    const float* wf  = (const float*)d_in[24]; const float* bf  = (const float*)d_in[25];

    float* out_fe8 = (float*)d_out;                          // (B, IN_DIM, H, W)
    float* out_fi2 = out_fe8 + (size_t)B_ * INDIM * HW;      // (B, C, H, W)

    // Carve workspace (256B aligned regions).
    char* wp = (char*)d_ws;
    auto carve = [&](size_t bytes) -> char* {
        char* p = wp; wp += (bytes + 255) & ~(size_t)255; return p;
    };
    const size_t NCH = (size_t)B_ * C_ * HW;   // 2,097,152 elements
    float*    x1F   = (float*)carve(NCH * 4);
    float*    x2F   = (float*)carve(NCH * 4);
    float*    yF    = (float*)carve(NCH * 4);      // conv1x1 pre-LN, token-major
    float*    y6aF  = (float*)carve(NCH * 4);      // NCHW, feeds conv3x3
    float*    zF    = (float*)carve(NCH * 4);
    float*    stats = (float*)carve(4 * B_ * 2 * sizeof(float));
    _Float16* fe1H  = (_Float16*)carve(NCH * 2);   // NCHW (= V^T layout)
    _Float16* fi2Hn = (_Float16*)carve(NCH * 2);   // NCHW
    _Float16* fi2Ht = (_Float16*)carve(NCH * 2);   // token-major
    _Float16* fi3T  = (_Float16*)carve(NCH * 2);   // token-major [b][m][c]
    _Float16* fe4H  = (_Float16*)carve(NCH * 2);   // token-major (Q)
    _Float16* aoH   = (_Float16*)carve(NCH * 2);   // attention out, token-major
    _Float16* fe5H  = (_Float16*)carve(NCH * 2);
    _Float16* fi6Ht = (_Float16*)carve(NCH * 2);
    _Float16* prodH = (_Float16*)carve(NCH * 2);
    _Float16* W3H   = (_Float16*)carve((size_t)HW * HW * 2);
    _Float16* W4H   = (_Float16*)carve((size_t)C_ * C_ * 2);
    _Float16* W5H   = (_Float16*)carve((size_t)C_ * C_ * 2);
    _Float16* w6aH  = (_Float16*)carve((size_t)C_ * C_ * 2);
    _Float16* wfH   = (_Float16*)carve((size_t)INDIM * C_ * 2);
    float* st1 = stats;           float* st2 = stats + 2 * B_;
    float* st3 = stats + 4 * B_;  float* st4 = stats + 6 * B_;

    dim3 blk(256);
    const int gElem = (int)(NCH / 256);   // 8192

    // f_1 / f_2 front convs + LayerNorms
    conv3x3_kernel<<<gElem, blk, 0, stream>>>(f_e, w1, b1, x1F, INDIM, C_);
    conv3x3_kernel<<<gElem, blk, 0, stream>>>(f_i, w2, b2, x2F, INDIM, C_);
    ln_reduce_kernel<<<B_, 512, 0, stream>>>(x1F, st1);
    ln_reduce_kernel<<<B_, 512, 0, stream>>>(x2F, st2);
    ln_apply_nchw_kernel<<<gElem, blk, 0, stream>>>(x1F, g1, be1, st1,
                                                    nullptr, fe1H, nullptr);
    ln_apply_nchw_kernel<<<gElem, blk, 0, stream>>>(x2F, g2, be2, st2,
                                                    out_fi2, fi2Hn, fi2Ht);

    // Weight conversions fp32 -> fp16
    cvt_f32_f16_kernel<<<(HW * HW) / 256, blk, 0, stream>>>(W3, W3H, HW * HW);
    cvt_f32_f16_kernel<<<(C_ * C_) / 256, blk, 0, stream>>>(W4, W4H, C_ * C_);
    cvt_f32_f16_kernel<<<(C_ * C_) / 256, blk, 0, stream>>>(W5, W5H, C_ * C_);
    cvt_f32_f16_kernel<<<(C_ * C_) / 256, blk, 0, stream>>>(w6a, w6aH, C_ * C_);
    cvt_f32_f16_kernel<<<(INDIM * C_) / 256, blk, 0, stream>>>(wf, wfH, INDIM * C_);

    // f_i_3 = lrelu(fi2(B,C,HW) . W3^T + b3)   M=512, N=4096, K=4096
    // out_mode=2 writes directly in attention K-tile layout [b][m][c].
    gemm_wmma_kernel<<<dim3(4096 / 64, 512 / 128), blk, 0, stream>>>(
        fi2Hn, W3H, b3, nullptr, fi3T, 512, 4096, 4096, 1, 2);
    // f_e_4 = lrelu(tok . W4^T + b4)           M=16384, N=128, K=128
    gemm_wmma_kernel<<<dim3(2, 128), blk, 0, stream>>>(
        fi2Ht, W4H, b4, nullptr, fe4H, B_ * HW, C_, C_, 1, 0);

    // attention: softmax(Q K^T) V, streamed (scores never hit HBM)
    flash_attn_kernel<<<dim3(HW / 64, B_), dim3(128), 0, stream>>>(
        fe4H, fi3T, fe1H, aoH);

    // f_e_5 = lrelu(attn_out . W5^T + b5)
    gemm_wmma_kernel<<<dim3(2, 128), blk, 0, stream>>>(
        aoH, W5H, b5, nullptr, fe5H, B_ * HW, C_, C_, 1, 0);

    // f_6 branch: conv1x1 (as GEMM) -> LN -> lrelu -> conv3x3 -> LN -> lrelu
    gemm_wmma_kernel<<<dim3(2, 128), blk, 0, stream>>>(
        fi2Ht, w6aH, b6a, yF, nullptr, B_ * HW, C_, C_, 0, 0);
    ln_reduce_kernel<<<B_, 512, 0, stream>>>(yF, st3);
    ln_apply_tok_kernel<<<gElem, blk, 0, stream>>>(yF, g6a, be6a, st3, y6aF);
    conv3x3_kernel<<<gElem, blk, 0, stream>>>(y6aF, w6b, b6b, zF, C_, C_);
    ln_reduce_kernel<<<B_, 512, 0, stream>>>(zF, st4);
    ln_apply_nchw_kernel<<<gElem, blk, 0, stream>>>(zF, g6b, be6b, st4,
                                                    nullptr, nullptr, fi6Ht);

    // f_e_7 = f_e_5 * f_i_6 ; f_e_8 = conv1x1(wf) -> fp32 NCHW into d_out
    mul_h_kernel<<<gElem, blk, 0, stream>>>(fe5H, fi6Ht, prodH);
    gemm_wmma_kernel<<<dim3(1, 128), blk, 0, stream>>>(
        prodH, wfH, bf, out_fe8, nullptr, B_ * HW, INDIM, C_, 0, 1);
}
